// DEQTrellisNetLM_76673756168215
// MI455X (gfx1250) — matrix operand order repaired and checked by reading, not compiled, not run on hardware
//
#include <hip/hip_runtime.h>
#include <hip/hip_bf16.h>

// ---------------------------------------------------------------------------
// DEQ-TrellisNet on gfx1250: bf16 WMMA GEMMs (f32 accum) + fused LSTM gating,
// with double-buffered LDS staging via global_load_async_to_lds_b128.
//   B=4, NINP=512, H=768, 4H=3072, K=2, L=1024, NLAYER=10, NOUT=256
// Activations stored shifted: buf[b][pos][C], pos = t+1, pos0 = z0 (constant),
// so every K=2 conv GEMM column (concat(h[t-1],h[t])) is contiguous.
// ---------------------------------------------------------------------------

typedef __attribute__((ext_vector_type(16))) __bf16 v16bf;
typedef __attribute__((ext_vector_type(8)))  float  v8f;

#define HC    768
#define LC    1024
#define BC    4
#define M4    3072
#define POS   1025      /* L+1 positions (pos 0 = z0 pad) */
#define KINJ  1024      /* 2*NINP */
#define KLAY  1536      /* 2*H    */
#define NOUTC 256
#define OUT0  1048576   /* B*L*NOUT, start of z0_out in d_out */
#define LROW  40        /* LDS row stride in elements: 80B = 5*16B, bank-safe */

union BFrag { v16bf v; uint4 q[2]; };
union AccU  { v8f v; float4 f[2]; float s[8]; };

__device__ __forceinline__ unsigned short f2bf(float f) {
  unsigned int u = __float_as_uint(f);
  u += 0x7FFFu + ((u >> 16) & 1u);           // round-to-nearest-even
  return (unsigned short)(u >> 16);
}
__device__ __forceinline__ float sigm(float x) { return 1.0f / (1.0f + __expf(-x)); }
__device__ __forceinline__ float tanh_f(float x) {
  x = fminf(fmaxf(x, -15.0f), 15.0f);
  float e = __expf(2.0f * x);
  return (e - 1.0f) / (e + 1.0f);
}

// CDNA5 async copy global -> LDS (ASYNCcnt); inline asm = toolchain-portable.
__device__ __forceinline__ void async_b128(const unsigned short* g, unsigned ldsByte) {
  asm volatile("global_load_async_to_lds_b128 %0, %1, off"
               :: "v"(ldsByte), "v"((unsigned long long)(uintptr_t)g)
               : "memory");
}
__device__ __forceinline__ void wait_async4() { asm volatile("s_wait_asynccnt 0x4" ::: "memory"); }
__device__ __forceinline__ void wait_async0() { asm volatile("s_wait_asynccnt 0x0" ::: "memory"); }

// ---- weight / input conversion --------------------------------------------
__global__ void cvt_wi(const float* __restrict__ w, unsigned short* __restrict__ o) {
  int idx = blockIdx.x * blockDim.x + threadIdx.x;
  if (idx >= M4 * KINJ) return;
  int row = idx >> 10, rem = idx & 1023;
  int kk = rem >> 9, i = rem & 511;
  o[idx] = f2bf(w[(row * 512 + i) * 2 + kk]);
}
__global__ void cvt_wc(const float* __restrict__ w, unsigned short* __restrict__ o) {
  int idx = blockIdx.x * blockDim.x + threadIdx.x;
  if (idx >= M4 * KLAY) return;
  int row = idx / KLAY, rem = idx - row * KLAY;
  int kk = rem / HC, i = rem - kk * HC;
  o[idx] = f2bf(w[(row * HC + i) * 2 + kk]);
}
__global__ void cvt_x(const float* __restrict__ X, unsigned short* __restrict__ o) {
  int idx = blockIdx.x * blockDim.x + threadIdx.x;
  if (idx >= BC * POS * 512) return;
  int b = idx / (POS * 512), rem = idx - b * (POS * 512);
  int pos = rem >> 9, i = rem & 511;
  float v = (pos == 0) ? 0.0f : X[((size_t)(b * 512 + i)) * LC + (pos - 1)];
  o[idx] = f2bf(v);
}
__global__ void init_state(const float* __restrict__ z0,
                           unsigned short* __restrict__ HA, unsigned short* __restrict__ HB,
                           float* __restrict__ CA, float* __restrict__ CB) {
  int idx = blockIdx.x * blockDim.x + threadIdx.x;
  if (idx >= BC * POS * HC) return;
  int b = idx / (POS * HC), rem = idx - b * (POS * HC);
  int pos = rem / HC, j = rem - pos * HC;
  float h0 = (pos == 0) ? z0[b * 1536 + j] : 0.0f;
  float c0 = (pos == 0) ? z0[b * 1536 + HC + j] : 0.0f;
  unsigned short hb = f2bf(h0);
  HA[idx] = hb; HB[idx] = hb; CA[idx] = c0; CB[idx] = c0;
}

// ---- shared WMMA mainloop: 128(M:4 gates x 32) x 128(N) tile, K-step 32 ----
// W   : bf16 weights, row-major [gate*HC + j][KDIM]
// Act : bf16 activations [B][POS][actw]; column (b,t) = KDIM contiguous elems
template<int KDIM>
__device__ __forceinline__ void mma_mainloop(const unsigned short* __restrict__ W,
                                             int jblk,
                                             const unsigned short* __restrict__ Act,
                                             int actw, int cblk, AccU* acc) {
  __shared__ unsigned short As[2][128 * LROW];   // 2 x 10 KB
  __shared__ unsigned short Bs[2][128 * LROW];

  const int tid  = threadIdx.x;
  const int lane = tid & 31;
  const int wv   = tid >> 5;
  const int wm = wv >> 2, wn = wv & 3;
  const int m16 = lane & 15, half = lane >> 4;
  constexpr int NK = KDIM / 32;

  // Fill assignment: 512 A-chunks + 512 B-chunks of 16B; 4 per thread/stage.
  const int q0  = (tid & 3) * 8;                 // K sub-offset (elements)
  const int rA0 = tid >> 2, rA1 = rA0 + 64;      // A tile rows [0,128)
  const int cB0 = tid >> 2, cB1 = cB0 + 64;      // B tile cols [0,128)

  const unsigned short* gA0 = W + (size_t)((rA0 >> 5) * HC + jblk + (rA0 & 31)) * KDIM + q0;
  const unsigned short* gA1 = W + (size_t)((rA1 >> 5) * HC + jblk + (rA1 & 31)) * KDIM + q0;
  const int colg0 = cblk + cB0, colg1 = cblk + cB1;
  const unsigned short* gB0 = Act + (size_t)((colg0 >> 10) * POS + (colg0 & 1023)) * actw + q0;
  const unsigned short* gB1 = Act + (size_t)((colg1 >> 10) * POS + (colg1 & 1023)) * actw + q0;

  const unsigned baseA = (unsigned)(uintptr_t)&As[0][0];
  const unsigned baseB = (unsigned)(uintptr_t)&Bs[0][0];
  const unsigned lA0 = (unsigned)(rA0 * LROW + q0) * 2u;
  const unsigned lA1 = (unsigned)(rA1 * LROW + q0) * 2u;
  const unsigned lB0 = (unsigned)(cB0 * LROW + q0) * 2u;
  const unsigned lB1 = (unsigned)(cB1 * LROW + q0) * 2u;
  const unsigned stageBytes = 128u * LROW * 2u;

  auto issue = [&](int stage, int k0) {
    unsigned sa = baseA + (unsigned)(stage & 1) * stageBytes;
    unsigned sb = baseB + (unsigned)(stage & 1) * stageBytes;
    async_b128(gA0 + k0, sa + lA0);
    async_b128(gA1 + k0, sa + lA1);
    async_b128(gB0 + k0, sb + lB0);
    async_b128(gB1 + k0, sb + lB1);
  };

  issue(0, 0);
  for (int ks = 0; ks < NK; ++ks) {
    if (ks + 1 < NK) { issue(ks + 1, (ks + 1) * 32); wait_async4(); }
    else             { wait_async0(); }
    __syncthreads();                       // stage ks visible to all waves

    const unsigned short* as = &As[ks & 1][0];
    const unsigned short* bs = &Bs[ks & 1][0];
    BFrag A[4], Bf[2];
#pragma unroll
    for (int g = 0; g < 4; ++g) {          // A frag: lane row m16, K 8*half/+16
      int row = g * 32 + wm * 16 + m16;
      A[g].q[0] = *(const uint4*)(as + row * LROW + 8 * half);
      A[g].q[1] = *(const uint4*)(as + row * LROW + 16 + 8 * half);
    }
#pragma unroll
    for (int cn = 0; cn < 2; ++cn) {       // B frag: lane col m16, K 8*half/+16
      int ci = wn * 32 + cn * 16 + m16;
      Bf[cn].q[0] = *(const uint4*)(bs + ci * LROW + 8 * half);
      Bf[cn].q[1] = *(const uint4*)(bs + ci * LROW + 16 + 8 * half);
    }
#pragma unroll
    for (int g = 0; g < 4; ++g)
#pragma unroll
      for (int cn = 0; cn < 2; ++cn)
        acc[g * 2 + cn].v = __builtin_amdgcn_wmma_f32_16x16x32_bf16(
            false, A[g].v, false, Bf[cn].v, (short)0, acc[g * 2 + cn].v, false, false);
    __syncthreads();                       // reads done before buffer reuse
  }
}

// ---- inject GEMM: us[col][m] = Wi @ Xcat(col) + inject_b + conv_b ---------
__global__ __launch_bounds__(256)
void gemm_inject(const unsigned short* __restrict__ Wi,
                 const unsigned short* __restrict__ Xb,
                 const float* __restrict__ bi, const float* __restrict__ bc,
                 float* __restrict__ us) {
  const int lane = threadIdx.x & 31, wv = threadIdx.x >> 5;
  const int wm = wv >> 2, wn = wv & 3;
  const int m16 = lane & 15, half = lane >> 4;
  const int jblk = blockIdx.x * 32;
  const int cblk = blockIdx.y * 128;
  const int j0 = jblk + wm * 16;
  const int c0 = cblk + wn * 32;

  AccU acc[8];
#pragma unroll
  for (int g = 0; g < 4; ++g) {
    int mb = g * HC + j0 + 8 * half;
    float4 b0 = *(const float4*)(bi + mb), b1 = *(const float4*)(bi + mb + 4);
    float4 d0 = *(const float4*)(bc + mb), d1 = *(const float4*)(bc + mb + 4);
    float4 s0 = make_float4(b0.x + d0.x, b0.y + d0.y, b0.z + d0.z, b0.w + d0.w);
    float4 s1 = make_float4(b1.x + d1.x, b1.y + d1.y, b1.z + d1.z, b1.w + d1.w);
    acc[g * 2 + 0].f[0] = s0; acc[g * 2 + 0].f[1] = s1;
    acc[g * 2 + 1].f[0] = s0; acc[g * 2 + 1].f[1] = s1;
  }

  mma_mainloop<KINJ>(Wi, jblk, Xb, 512, cblk, acc);

#pragma unroll
  for (int g = 0; g < 4; ++g)
#pragma unroll
    for (int cn = 0; cn < 2; ++cn) {
      int col = c0 + cn * 16 + m16;
      float* up = us + (size_t)col * M4 + g * HC + j0 + 8 * half;
      *(float4*)up = acc[g * 2 + cn].f[0];
      *(float4*)(up + 4) = acc[g * 2 + cn].f[1];
    }
}

// ---- layer GEMM + fused LSTM gating ---------------------------------------
__global__ __launch_bounds__(256)
void gemm_layer(const unsigned short* __restrict__ Wc,
                const unsigned short* __restrict__ Hp,  // prev h, bf16 [B][POS][H]
                const float* __restrict__ Cp,           // prev c, f32 [B][POS][H]
                const float* __restrict__ us,           // [4096][3072]
                unsigned short* __restrict__ Hn,
                float* __restrict__ Cn,
                float* __restrict__ dout, int last) {
  const int lane = threadIdx.x & 31, wv = threadIdx.x >> 5;
  const int wm = wv >> 2, wn = wv & 3;
  const int m16 = lane & 15, half = lane >> 4;
  const int jblk = blockIdx.x * 32;
  const int cblk = blockIdx.y * 128;
  const int j0 = jblk + wm * 16;
  const int c0 = cblk + wn * 32;

  AccU acc[8];
#pragma unroll
  for (int g = 0; g < 4; ++g)
#pragma unroll
    for (int cn = 0; cn < 2; ++cn) {
      int col = c0 + cn * 16 + m16;
      const float* up = us + (size_t)col * M4 + g * HC + j0 + 8 * half;
      acc[g * 2 + cn].f[0] = *(const float4*)up;
      acc[g * 2 + cn].f[1] = *(const float4*)(up + 4);
    }

  mma_mainloop<KLAY>(Wc, jblk, Hp, HC, cblk, acc);

  // LSTM gating: gates for (j,col) sit in the same lane/slot across gate tiles.
#pragma unroll
  for (int cn = 0; cn < 2; ++cn) {
    int col = c0 + cn * 16 + m16;
    int b = col >> 10, t = col & 1023;
    const float* cpp = Cp + (size_t)(b * POS + t) * HC;   // c_prev at t-1 (shifted)
    size_t nb = (size_t)(b * POS + t + 1) * HC;
#pragma unroll
    for (int r = 0; r < 8; ++r) {
      int j = j0 + 8 * half + r;
      float it = acc[0 + cn].s[r];
      float ot = acc[2 + cn].s[r];
      float gt = acc[4 + cn].s[r];
      float ft = acc[6 + cn].s[r];
      float cp = cpp[j];
      float ct = sigm(ft) * cp + sigm(it) * tanh_f(gt);
      float ht = sigm(ot) * tanh_f(ct);
      Cn[nb + j] = ct;
      Hn[nb + j] = f2bf(ht);
      if (last) {
        if (j >= HC - NOUTC)                      // out[b][t][n], n = j-512
          dout[(size_t)col * NOUTC + (j - (HC - NOUTC))] = ht;
        if (t == LC - 1) {                        // z0_out = [h; c] at t = L-1
          dout[OUT0 + b * 1536 + j]      = ht;
          dout[OUT0 + b * 1536 + HC + j] = ct;
        }
      }
    }
  }
}

// ---------------------------------------------------------------------------
extern "C" void kernel_launch(void* const* d_in, const int* in_sizes, int n_in,
                              void* d_out, int out_size, void* d_ws, size_t ws_size,
                              hipStream_t stream) {
  (void)in_sizes; (void)n_in; (void)out_size; (void)ws_size;
  const float* X  = (const float*)d_in[0];
  const float* z0 = (const float*)d_in[1];
  const float* iw = (const float*)d_in[2];
  const float* ib = (const float*)d_in[3];
  const float* cw = (const float*)d_in[4];
  const float* cb = (const float*)d_in[5];
  float* out = (float*)d_out;

  char* p = (char*)d_ws;
  float* us           = (float*)p;          p += (size_t)4096 * 3072 * 4;  // 50.3 MB
  unsigned short* WiB = (unsigned short*)p; p += (size_t)3072 * 1024 * 2;  //  6.3 MB
  unsigned short* WcB = (unsigned short*)p; p += (size_t)3072 * 1536 * 2;  //  9.4 MB
  unsigned short* Xb  = (unsigned short*)p; p += (size_t)BC * POS * 512 * 2;
  unsigned short* HA  = (unsigned short*)p; p += (size_t)BC * POS * HC * 2;
  unsigned short* HB  = (unsigned short*)p; p += (size_t)BC * POS * HC * 2;
  float* CA           = (float*)p;          p += (size_t)BC * POS * HC * 4;
  float* CB           = (float*)p;

  cvt_wi<<<(M4 * KINJ + 255) / 256, 256, 0, stream>>>(iw, WiB);
  cvt_wc<<<(M4 * KLAY + 255) / 256, 256, 0, stream>>>(cw, WcB);
  cvt_x <<<(BC * POS * 512 + 255) / 256, 256, 0, stream>>>(X, Xb);
  init_state<<<(BC * POS * HC + 255) / 256, 256, 0, stream>>>(z0, HA, HB, CA, CB);

  dim3 grid(24, 32);  // j-blocks (768/32) x col-blocks (4096/128)
  gemm_inject<<<grid, 256, 0, stream>>>(WiB, Xb, ib, cb, us);

  unsigned short* Hbuf[2] = {HA, HB};
  float*          Cbuf[2] = {CA, CB};
  for (int i = 0; i < 10; ++i) {
    gemm_layer<<<grid, 256, 0, stream>>>(WcB, Hbuf[i & 1], Cbuf[i & 1], us,
                                         Hbuf[(i + 1) & 1], Cbuf[(i + 1) & 1],
                                         out, i == 9);
  }
}
